// VNNGP_11905649344985
// MI455X (gfx1250) — compile-verified, matrix operand order checked by reading, not run on hardware
//
#include <hip/hip_runtime.h>
#include <hip/hip_bf16.h>
#include <math.h>

#define Mdim   1024
#define Ndim   65536
#define DIMK   8
#define JITTER 1e-4f

typedef __attribute__((ext_vector_type(2)))  float    v2f;
typedef __attribute__((ext_vector_type(8)))  float    v8f;
typedef __attribute__((ext_vector_type(16))) _Float16 v16h;

#if __has_builtin(__builtin_amdgcn_wmma_f32_16x16x4_f32)
#define HAVE_WMMA_F32X4 1
#else
#define HAVE_WMMA_F32X4 0
#endif

// gfx1250 async global->LDS path (ASYNCcnt-tracked, bypasses VGPRs).
// Probe (round 2 diagnostic) showed the builtin takes pointers to a 16-byte
// int vector: AS1 (global) source, AS3 (LDS) destination.
#if defined(__AMDGCN__) &&                                         \
    __has_builtin(__builtin_amdgcn_global_load_async_to_lds_b128) && \
    __has_builtin(__builtin_amdgcn_s_wait_asynccnt)
#define HAVE_ASYNC_LDS 1
typedef int async_v4i __attribute__((vector_size(16)));
typedef __attribute__((address_space(1))) async_v4i gbl_v4i;
typedef __attribute__((address_space(3))) async_v4i lds_v4i;
#else
#define HAVE_ASYNC_LDS 0
#endif

// Per-wave 16x16 f32 tile accumulation: C += Arows * Brows^T, where Arow/Brow
// point at row (tileBase + lane%16) of a row-major matrix and KTOT is the
// contraction depth (multiple of 4).
// f32 path: V_WMMA_F32_16X16X4_F32 layout:
//   A 16x4: lane%16 = M row, lane/16 selects K pair {0,1} vs {2,3}
//   B 4x16: lane%16 = N col, lane/16 selects K pair (B fed as rows of the
//           transposed operand, i.e. B[k][n] = Brow_n[k])
//   C 16x16: VGPR v holds M = v + 8*(lane/16), N = lane%16
__device__ __forceinline__ v8f wmma_rowrow(const float* __restrict__ Arow,
                                           const float* __restrict__ Brow,
                                           int KTOT, int hi, v8f c)
{
#if HAVE_WMMA_F32X4
    for (int kk = 0; kk < KTOT; kk += 4) {
        int kb = kk + (hi << 1);
        v2f a, b;
        a.x = Arow[kb]; a.y = Arow[kb + 1];
        b.x = Brow[kb]; b.y = Brow[kb + 1];
        c = __builtin_amdgcn_wmma_f32_16x16x4_f32(false, a, false, b,
                                                  (short)0, c, false, false);
    }
#else
    // Fallback: confirmed f16 WMMA, zero-padded K (16-bit A 16x32 layout).
    for (int kk = 0; kk < KTOT; kk += 32) {
        v16h a, b;
#pragma unroll
        for (int e = 0; e < 16; ++e) {
            int vg = e >> 1, lo = e & 1;
            int k = (vg < 4) ? (kk + hi * 8 + vg * 2 + lo)
                             : (kk + 16 + hi * 8 + (vg - 4) * 2 + lo);
            a[e] = (k < KTOT) ? (_Float16)Arow[k] : (_Float16)0.0f;
            b[e] = (k < KTOT) ? (_Float16)Brow[k] : (_Float16)0.0f;
        }
        c = __builtin_amdgcn_wmma_f32_16x16x32_f16(false, a, false, b,
                                                   (short)0, c, false, false);
    }
#endif
    return c;
}

// ---------------------------------------------------------------------------
// 1) Lu = tril(Lu_raw,-1) + diag(exp(diag(Lu_raw)));  zz[j] = |Z_j|^2;  mu copy
// ---------------------------------------------------------------------------
__global__ void setup_kernel(const float* __restrict__ Z,
                             const float* __restrict__ Lu_raw,
                             const float* __restrict__ mu,
                             float* __restrict__ zz,
                             float* __restrict__ outLu,
                             float* __restrict__ outMu)
{
    int id = blockIdx.x * blockDim.x + threadIdx.x;
    if (id < Mdim * Mdim) {
        int i = id / Mdim, j = id - i * Mdim;
        float v = Lu_raw[id];
        outLu[id] = (j < i) ? v : ((j == i) ? expf(v) : 0.0f);
    }
    if (id < Mdim) {
        float s = 0.0f;
#pragma unroll
        for (int d = 0; d < DIMK; ++d) {
            float zd = Z[id * DIMK + d];
            s = fmaf(zd, zd, s);
        }
        zz[id]    = s;
        outMu[id] = mu[id];
    }
}

// ---------------------------------------------------------------------------
// 2) Kzz_j = var*exp(-0.5*sqdist(Z,Z)/ls^2) + JITTER*I  (WMMA Z@Z^T)
//    Written to ws (for per-point gathers) and to outL (Cholesky workspace).
// ---------------------------------------------------------------------------
__global__ void __launch_bounds__(256)
kzz_kernel(const float* __restrict__ Z, const float* __restrict__ zz,
           const float* __restrict__ lsp, const float* __restrict__ varp,
           float* __restrict__ Kzz, float* __restrict__ outL)
{
    int wave = blockIdx.x * (blockDim.x >> 5) + (threadIdx.x >> 5);
    int lane = threadIdx.x & 31;
    int r = lane & 15, hi = lane >> 4;
    int ti = (wave >> 6) << 4;   // 64 tiles per side
    int tj = (wave & 63) << 4;

    v8f c = {};
    c = wmma_rowrow(Z + (ti + r) * DIMK, Z + (tj + r) * DIMK, DIMK, hi, c);

    float ls   = lsp[0];
    float var  = varp[0];
    float sc   = -0.5f / (ls * ls);
    int   col  = tj + r;
    float zzc  = zz[col];
    int   rb   = ti + (hi << 3);
#pragma unroll
    for (int v = 0; v < 8; ++v) {
        int   row = rb + v;
        float d2  = fmaxf(zz[row] + zzc - 2.0f * c[v], 0.0f);
        float val = var * expf(sc * d2);
        if (row == col) val += JITTER;
        Kzz[row * Mdim + col]  = val;
        outL[row * Mdim + col] = val;
    }
}

// ---------------------------------------------------------------------------
// 3) S = Lu @ Lu^T   (WMMA, K = 1024)
// ---------------------------------------------------------------------------
__global__ void __launch_bounds__(256)
s_gemm_kernel(const float* __restrict__ Lu, float* __restrict__ S)
{
    int wave = blockIdx.x * (blockDim.x >> 5) + (threadIdx.x >> 5);
    int lane = threadIdx.x & 31;
    int r = lane & 15, hi = lane >> 4;
    int ti = (wave >> 6) << 4;
    int tj = (wave & 63) << 4;

    v8f c = {};
    c = wmma_rowrow(Lu + (ti + r) * Mdim, Lu + (tj + r) * Mdim, Mdim, hi, c);

    int col = tj + r;
    int rb  = ti + (hi << 3);
#pragma unroll
    for (int v = 0; v < 8; ++v)
        S[(rb + v) * Mdim + col] = c[v];
}

// ---------------------------------------------------------------------------
// 4) In-place lower Cholesky of outL (single workgroup, row-per-thread rank-1)
// ---------------------------------------------------------------------------
__global__ void __launch_bounds__(1024)
cholesky_kernel(float* __restrict__ A)
{
    int tid = threadIdx.x;
    for (int k = 0; k < Mdim; ++k) {
        __syncthreads();
        if (tid == 0) A[k * Mdim + k] = sqrtf(A[k * Mdim + k]);
        __syncthreads();
        float invd = 1.0f / A[k * Mdim + k];
        int i = k + 1 + tid;                 // 1024 threads cover all rows
        if (i < Mdim) A[i * Mdim + k] *= invd;
        __syncthreads();
        if (i < Mdim) {
            float lik = A[i * Mdim + k];
            for (int j = k + 1; j <= i; ++j)
                A[i * Mdim + j] = fmaf(-lik, A[j * Mdim + k], A[i * Mdim + j]);
        }
    }
    __syncthreads();
    for (int id = tid; id < Mdim * Mdim; id += blockDim.x) {
        int i = id / Mdim, j = id - i * Mdim;
        if (j > i) A[id] = 0.0f;
    }
}

// ---------------------------------------------------------------------------
// 5) Fused: distances to all Z (Z staged in LDS via async global->LDS copies),
//    top-3 select, 3x3 inverse, W, mean, quadratic forms, scale.
// ---------------------------------------------------------------------------
__global__ void __launch_bounds__(256)
predict_kernel(const float* __restrict__ X, const float* __restrict__ Z,
               const float* __restrict__ mu, const float* __restrict__ zz,
               const float* __restrict__ Kzz, const float* __restrict__ S,
               const float* __restrict__ lsp, const float* __restrict__ varp,
               float* __restrict__ meanOut, float* __restrict__ scaleOut)
{
    __shared__ float4 sZ4[Mdim * 2];      // 32 KB: Z rows (8 floats = 2x float4)
    __shared__ float4 sZZ4[Mdim / 4];     //  4 KB: |z|^2

    int tid = threadIdx.x;
    float* sZZ = (float*)sZZ4;

#if HAVE_ASYNC_LDS
    {
        // GLOBAL_LOAD_ASYNC_TO_LDS_B128: each lane copies 16B straight into
        // LDS without touching VGPRs; one instruction moves 512B per wave.
        const float4* Zg  = (const float4*)Z;
        const float4* zzg = (const float4*)zz;
        for (int i = tid; i < Mdim * 2; i += blockDim.x)
            __builtin_amdgcn_global_load_async_to_lds_b128(
                (gbl_v4i*)(Zg + i), (lds_v4i*)(sZ4 + i), 0, 0);
        for (int i = tid; i < Mdim / 4; i += blockDim.x)
            __builtin_amdgcn_global_load_async_to_lds_b128(
                (gbl_v4i*)(zzg + i), (lds_v4i*)(sZZ4 + i), 0, 0);
        __builtin_amdgcn_s_wait_asynccnt(0);   // this wave's copies done
    }
#else
    {
        float* sZf = (float*)sZ4;
        for (int i = tid; i < Mdim * DIMK; i += blockDim.x) sZf[i] = Z[i];
        for (int i = tid; i < Mdim; i += blockDim.x) sZZ[i] = zz[i];
    }
#endif
    __syncthreads();                           // all waves' copies visible

    int n = blockIdx.x * blockDim.x + tid;
    if (n >= Ndim) return;

    const float4* x4 = (const float4*)(X + n * DIMK);
    float4 xa = x4[0], xb = x4[1];
    float xx = xa.x * xa.x + xa.y * xa.y + xa.z * xa.z + xa.w * xa.w +
               xb.x * xb.x + xb.y * xb.y + xb.z * xb.z + xb.w * xb.w;

    float bd0 = 3.4e38f, bd1 = 3.4e38f, bd2 = 3.4e38f;
    int   bi0 = 0,       bi1 = 1,       bi2 = 2;

    for (int j = 0; j < Mdim; ++j) {
        float4 za = sZ4[2 * j], zb = sZ4[2 * j + 1];
        float dot = xa.x * za.x;
        dot = fmaf(xa.y, za.y, dot); dot = fmaf(xa.z, za.z, dot);
        dot = fmaf(xa.w, za.w, dot); dot = fmaf(xb.x, zb.x, dot);
        dot = fmaf(xb.y, zb.y, dot); dot = fmaf(xb.z, zb.z, dot);
        dot = fmaf(xb.w, zb.w, dot);
        float d2 = fmaxf(xx + sZZ[j] - 2.0f * dot, 0.0f);
        if (d2 < bd0) {
            bd2 = bd1; bi2 = bi1; bd1 = bd0; bi1 = bi0; bd0 = d2; bi0 = j;
        } else if (d2 < bd1) {
            bd2 = bd1; bi2 = bi1; bd1 = d2; bi1 = j;
        } else if (d2 < bd2) {
            bd2 = d2; bi2 = j;
        }
    }

    float ls  = lsp[0];
    float var = varp[0];
    float sc  = -0.5f / (ls * ls);

    float kxz[3];
    kxz[0] = var * expf(sc * bd0);
    kxz[1] = var * expf(sc * bd1);
    kxz[2] = var * expf(sc * bd2);
    int bi[3] = { bi0, bi1, bi2 };

    // Gather 3x3 little_Kzz (from jittered Kzz) and little_S
    float G[9], Sv[9];
#pragma unroll
    for (int a = 0; a < 3; ++a) {
        const float* Kr = Kzz + bi[a] * Mdim;
        const float* Sr = S   + bi[a] * Mdim;
#pragma unroll
        for (int b = 0; b < 3; ++b) {
            G[a * 3 + b]  = Kr[bi[b]];
            Sv[a * 3 + b] = Sr[bi[b]];
        }
    }

    // Invert (G + JITTER*I) via adjugate
    float m00 = G[0] + JITTER, m01 = G[1],          m02 = G[2];
    float m10 = G[3],          m11 = G[4] + JITTER, m12 = G[5];
    float m20 = G[6],          m21 = G[7],          m22 = G[8] + JITTER;
    float i00 = m11 * m22 - m12 * m21;
    float i01 = m02 * m21 - m01 * m22;
    float i02 = m01 * m12 - m02 * m11;
    float i10 = m12 * m20 - m10 * m22;
    float i11 = m00 * m22 - m02 * m20;
    float i12 = m02 * m10 - m00 * m12;
    float i20 = m10 * m21 - m11 * m20;
    float i21 = m01 * m20 - m00 * m21;
    float i22 = m00 * m11 - m01 * m10;
    float invdet = 1.0f / (m00 * i00 + m01 * i10 + m02 * i20);

    float W[3];
    W[0] = (kxz[0] * i00 + kxz[1] * i10 + kxz[2] * i20) * invdet;
    W[1] = (kxz[0] * i01 + kxz[1] * i11 + kxz[2] * i21) * invdet;
    W[2] = (kxz[0] * i02 + kxz[1] * i12 + kxz[2] * i22) * invdet;

    float mean = W[0] * mu[bi[0]] + W[1] * mu[bi[1]] + W[2] * mu[bi[2]];

    float qG = 0.0f, qS = 0.0f;
#pragma unroll
    for (int a = 0; a < 3; ++a)
#pragma unroll
        for (int b = 0; b < 3; ++b) {
            qG = fmaf(W[a] * W[b], G[a * 3 + b],  qG);
            qS = fmaf(W[a] * W[b], Sv[a * 3 + b], qS);
        }

    float cov = var - qG + qS;
    meanOut[n]  = mean;
    scaleOut[n] = sqrtf(fmaxf(cov, 0.05f));
}

// ---------------------------------------------------------------------------
extern "C" void kernel_launch(void* const* d_in, const int* in_sizes, int n_in,
                              void* d_out, int out_size, void* d_ws, size_t ws_size,
                              hipStream_t stream)
{
    (void)in_sizes; (void)n_in; (void)out_size; (void)ws_size;

    const float* X      = (const float*)d_in[0];
    const float* Z      = (const float*)d_in[1];
    const float* Lu_raw = (const float*)d_in[2];
    const float* mu     = (const float*)d_in[3];
    const float* ls     = (const float*)d_in[4];
    const float* var    = (const float*)d_in[5];

    float* out      = (float*)d_out;
    float* outMean  = out;
    float* outScale = out + Ndim;
    float* outMu    = out + 2 * Ndim;
    float* outLu    = out + 2 * Ndim + Mdim;
    float* outL     = out + 2 * Ndim + Mdim + Mdim * Mdim;

    float* ws    = (float*)d_ws;
    float* wsKzz = ws;                      // M*M
    float* wsS   = ws + Mdim * Mdim;        // M*M
    float* wsZZ  = ws + 2 * Mdim * Mdim;    // M

    setup_kernel<<<(Mdim * Mdim) / 256, 256, 0, stream>>>(Z, Lu_raw, mu,
                                                          wsZZ, outLu, outMu);
    // 64x64 tiles of 16x16, 8 waves/block -> 512 blocks
    kzz_kernel<<<512, 256, 0, stream>>>(Z, wsZZ, ls, var, wsKzz, outL);
    s_gemm_kernel<<<512, 256, 0, stream>>>(outLu, wsS);
    cholesky_kernel<<<1, 1024, 0, stream>>>(outL);
    predict_kernel<<<Ndim / 256, 256, 0, stream>>>(X, Z, mu, wsZZ, wsKzz, wsS,
                                                   ls, var, outMean, outScale);
}
